// LTCCell_63702954934929
// MI455X (gfx1250) — matrix operand order repaired and checked by reading, not compiled, not run on hardware
//
#include <hip/hip_runtime.h>

// ---------------------------------------------------------------------------
// LTC cell, fused dual-GEMM + epilogue for gfx1250 (MI455X).
//   combined = [x | h]  (4096 x 3072)
//   f = tanh(combined @ W_mod + b_mod);  a = tanh(combined @ W_tgt + b_tgt)
//   tau = exp(log_tau); tau_sys = tau/(1+tau*|f|)
//   decay = exp(-dt / (tau_sys + 1e-6))
//   h_new = decay*h + (1-decay)*(a + bias)
//
// v3: native __bf16 converts (frees the WMMA-shadow VALU budget) and all
// A/B fragments hoisted to registers before the WMMA block so the 16
// matrix ops stream back-to-back behind a single s_wait_dscnt.
// ---------------------------------------------------------------------------

#define IN_SZ   1024
#define HID_SZ  2048
#define KTOT    3072
#define BATCHN  4096

#define BM 128
#define BN 128
#define BK 32
#define ASTR 40   // padded LDS row stride (bf16 elems) = 80B, 16B-aligned

typedef __attribute__((ext_vector_type(16))) __bf16          bf16x16;
typedef __attribute__((ext_vector_type(8)))  float           f32x8;
typedef __attribute__((ext_vector_type(4)))  float           f32x4;
typedef __attribute__((ext_vector_type(4)))  unsigned int    u32x4;
typedef __attribute__((ext_vector_type(4)))  unsigned short  u16x4;

union Frag16 { bf16x16 v; u32x4 q[2]; };

__device__ __forceinline__ unsigned short f2bf(float f) {
  // Native f32->bf16 convert (RNE); lowers to v_cvt*bf16* on gfx1250.
  return __builtin_bit_cast(unsigned short, static_cast<__bf16>(f));
}

// A fragment, 16x32 bf16 (ISA 7.12.2): lanes 0-15 -> row M=lane, K {0-7,16-23};
// lanes 16-31 -> row M=lane-16, K {8-15,24-31}. Tile is row-major in LDS.
__device__ __forceinline__ bf16x16 fragA(const unsigned short* tile, int lane) {
  const unsigned short* p = tile + (lane & 15) * ASTR + ((lane >> 4) << 3);
  Frag16 f;
  f.q[0] = *reinterpret_cast<const u32x4*>(p);        // K 0-7  / 8-15
  f.q[1] = *reinterpret_cast<const u32x4*>(p + 16);   // K 16-23 / 24-31
  return f.v;
}

// B fragment, 32x16 bf16: lanes 0-15 -> col N=lane, K {0-7,16-23};
// lanes 16-31 -> col N=lane-16, K {8-15,24-31}. Tile stored N-major
// (transposed) in LDS so this is two b128 reads per lane.
__device__ __forceinline__ bf16x16 fragB(const unsigned short* tile, int lane) {
  const unsigned short* p = tile + (lane & 15) * ASTR + ((lane >> 4) << 3);
  Frag16 f;
  f.q[0] = *reinterpret_cast<const u32x4*>(p);        // K 0-7  / 8-15
  f.q[1] = *reinterpret_cast<const u32x4*>(p + 16);   // K 16-23 / 24-31
  return f.v;
}

__device__ __forceinline__ f32x8 wmma_bf16(bf16x16 a, bf16x16 b, f32x8 c) {
  // (neg_a, A, neg_b, B, c_mod, C, reuse_a, reuse_b)
  return __builtin_amdgcn_wmma_f32_16x16x32_bf16(false, a, false, b,
                                                 (short)0, c, false, false);
}

__global__ __launch_bounds__(256)
void ltc_fused_wmma_bf16(const float* __restrict__ x,
                         const float* __restrict__ hin,
                         const float* __restrict__ td,
                         const float* __restrict__ Wm,
                         const float* __restrict__ bm,
                         const float* __restrict__ Wt,
                         const float* __restrict__ bt,
                         const float* __restrict__ logtau,
                         const float* __restrict__ bias,
                         float* __restrict__ out) {
  // Double-buffered tiles: 2*(10 + 2*10) KB = 60 KB LDS.
  __shared__ __align__(16) unsigned short Alds[2][BM * ASTR];
  __shared__ __align__(16) unsigned short Blds[2][2][BN * ASTR];

  const int tid   = threadIdx.x;
  const int lane  = tid & 31;
  const int wave  = tid >> 5;          // 0..7
  const int waveM = wave >> 2;         // 0..1 -> 64 rows each
  const int waveN = wave & 3;          // 0..3 -> 32 cols each
  const int m0 = blockIdx.y * BM;
  const int n0 = blockIdx.x * BN;

  // Per-thread fixed global-tile coordinates (only kb varies per iteration).
  const int aRow = tid >> 1;                 // 0..127
  const int aC4  = (tid & 1) << 4;           // 0 or 16 (+ i*4)
  const int bKr  = tid >> 3;                 // 0..31
  const int bC4  = (tid & 7) << 4;           // 0,16,..,112 (+ i*4)

  f32x8 acc[2][4][2];                  // [branch][mtile][ntile]
  const f32x8 zero = {0.f, 0.f, 0.f, 0.f, 0.f, 0.f, 0.f, 0.f};
#pragma unroll
  for (int b = 0; b < 2; ++b)
#pragma unroll
    for (int mt = 0; mt < 4; ++mt)
#pragma unroll
      for (int nt = 0; nt < 2; ++nt) acc[b][mt][nt] = zero;

  f32x4 aReg[4];          // A slab staging
  f32x4 bReg[2][4];       // B slab staging (per weight)

  auto load_global = [&](int kb) {
#pragma unroll
    for (int i = 0; i < 4; ++i) {
      const int gcol = kb + aC4 + (i << 2);
      if (gcol < IN_SZ)
        aReg[i] = *reinterpret_cast<const f32x4*>(
            x + (size_t)(m0 + aRow) * IN_SZ + gcol);
      else
        aReg[i] = *reinterpret_cast<const f32x4*>(
            hin + (size_t)(m0 + aRow) * HID_SZ + (gcol - IN_SZ));
    }
#pragma unroll
    for (int w = 0; w < 2; ++w) {
      const float* W = w ? Wt : Wm;
#pragma unroll
      for (int i = 0; i < 4; ++i)
        bReg[w][i] = *reinterpret_cast<const f32x4*>(
            W + (size_t)(kb + bKr) * HID_SZ + n0 + bC4 + (i << 2));
    }
  };

  auto store_lds = [&](int buf) {
#pragma unroll
    for (int i = 0; i < 4; ++i) {
      u16x4 s = { f2bf(aReg[i].x), f2bf(aReg[i].y),
                  f2bf(aReg[i].z), f2bf(aReg[i].w) };
      *reinterpret_cast<u16x4*>(&Alds[buf][aRow * ASTR + aC4 + (i << 2)]) = s;
    }
#pragma unroll
    for (int w = 0; w < 2; ++w) {
#pragma unroll
      for (int i = 0; i < 4; ++i) {
        const int n = bC4 + (i << 2);               // N within tile, transposed
        Blds[buf][w][(n + 0) * ASTR + bKr] = f2bf(bReg[w][i].x);
        Blds[buf][w][(n + 1) * ASTR + bKr] = f2bf(bReg[w][i].y);
        Blds[buf][w][(n + 2) * ASTR + bKr] = f2bf(bReg[w][i].z);
        Blds[buf][w][(n + 3) * ASTR + bKr] = f2bf(bReg[w][i].w);
      }
    }
  };

  // Prologue: fill buffer 0.
  load_global(0);
  store_lds(0);
  __syncthreads();

  const int nIter = KTOT / BK;          // 96
  for (int it = 0; it < nIter; ++it) {
    const int cur  = it & 1;
    const int nxt  = cur ^ 1;
    const bool more = (it + 1) < nIter;

    // Issue next slab's global loads first: v_wmma below overlaps latency.
    if (more) load_global((it + 1) * BK);

    // L2 prefetch two slabs ahead (global_prefetch_b8).
    if (it + 2 < nIter) {
      const int gcol = (it + 2) * BK + aC4;
      const float* pa = (gcol < IN_SZ)
          ? (x + (size_t)(m0 + aRow) * IN_SZ + gcol)
          : (hin + (size_t)(m0 + aRow) * HID_SZ + (gcol - IN_SZ));
      __builtin_prefetch(pa, 0, 1);
    }

    // ---- Hoist ALL fragments, then stream 16 WMMAs back-to-back ----
    bf16x16 aF[4];
    bf16x16 bF[2][2];
#pragma unroll
    for (int mt = 0; mt < 4; ++mt)
      aF[mt] = fragA(&Alds[cur][(waveM * 64 + mt * 16) * ASTR], lane);
#pragma unroll
    for (int w = 0; w < 2; ++w)
#pragma unroll
      for (int nt = 0; nt < 2; ++nt)
        bF[w][nt] = fragB(&Blds[cur][w][(waveN * 32 + nt * 16) * ASTR], lane);

#pragma unroll
    for (int w = 0; w < 2; ++w)
#pragma unroll
      for (int nt = 0; nt < 2; ++nt)
#pragma unroll
        for (int mt = 0; mt < 4; ++mt)
          acc[w][mt][nt] = wmma_bf16(aF[mt], bF[w][nt], acc[w][mt][nt]);

    // Convert + park next slab in the other buffer, then one barrier.
    if (more) store_lds(nxt);
    __syncthreads();
  }

  // ---- Epilogue straight out of accumulators ----
  // C/D layout: lane 0-15 -> N=lane, M=vgpr; lane 16-31 -> N=lane-16, M=vgpr+8.
#pragma unroll
  for (int mt = 0; mt < 4; ++mt) {
#pragma unroll
    for (int nt = 0; nt < 2; ++nt) {
      const int n   = n0 + waveN * 32 + nt * 16 + (lane & 15);
      const float bm_ = bm[n];
      const float bt_ = bt[n];
      const float bs_ = bias[n];
      const float tau = __expf(logtau[n]);
      const int mbase = m0 + waveM * 64 + mt * 16 + ((lane >> 4) << 3);
#pragma unroll
      for (int e = 0; e < 8; ++e) {
        const int m = mbase + e;
        const float f = tanhf(acc[0][mt][nt][e] + bm_);
        const float a = tanhf(acc[1][mt][nt][e] + bt_);
        const float tau_sys = tau / (1.0f + tau * fabsf(f));
        const float decay = __expf(-td[m] / (tau_sys + 1e-6f));
        const float hv = hin[(size_t)m * HID_SZ + n];
        out[(size_t)m * HID_SZ + n] = decay * hv + (1.0f - decay) * (a + bs_);
      }
    }
  }
}

extern "C" void kernel_launch(void* const* d_in, const int* in_sizes, int n_in,
                              void* d_out, int out_size, void* d_ws, size_t ws_size,
                              hipStream_t stream) {
  (void)in_sizes; (void)n_in; (void)out_size; (void)d_ws; (void)ws_size;
  const float* x      = (const float*)d_in[0];
  const float* h      = (const float*)d_in[1];
  const float* td     = (const float*)d_in[2];
  const float* W_mod  = (const float*)d_in[3];
  const float* b_mod  = (const float*)d_in[4];
  const float* W_tgt  = (const float*)d_in[5];
  const float* b_tgt  = (const float*)d_in[6];
  const float* logtau = (const float*)d_in[7];
  const float* bias   = (const float*)d_in[8];
  float* out = (float*)d_out;

  dim3 grid(HID_SZ / BN, BATCHN / BM);   // 16 x 32 = 512 blocks
  dim3 block(256);                       // 8 wave32
  ltc_fused_wmma_bf16<<<grid, block, 0, stream>>>(
      x, h, td, W_mod, b_mod, W_tgt, b_tgt, logtau, bias, out);
}
